// FunctionCorrelation_17282948399394
// MI455X (gfx1250) — matrix operand order repaired and checked by reading, not compiled, not run on hardware
//
#include <hip/hip_runtime.h>

// Correlation cost volume (FlowNet-style), MAX_DISP=4 -> 81 displacements.
// B=4, C=128, H=128, W=192.  out[b,(dy+4)*9+(dx+4),y,x] =
//   (1/C) * sum_c first[b,c,y,x] * second[b,c,y+dy,x+dx]   (zero padded)
//
// Strategy: per block, one 16-pixel x-tile of one (b,y).  Stage first tile
// (16px x 128ch) and second rows y-4..y+4 (9 x 24px x 128ch) into LDS as
// bf16, then 9 waves (one per dy) each run 16x16x32 bf16 WMMA chains over
// K=128, producing the 16x16 pixel-by-shifted-column products; the +/-4 dx
// band is covered by two overlapping column tiles (offsets 0 and 8).

typedef __bf16 bf16_t;
typedef __attribute__((ext_vector_type(16))) __bf16 v16bf;
typedef __attribute__((ext_vector_type(8)))  __bf16 v8bf;
typedef __attribute__((ext_vector_type(8)))  float  v8f;

#define BATCH 4
#define CCH   128
#define HH    128
#define WW    192
#define PDISP 4
#define NDY   9
#define NDISP 81
#define TPX   16              // output pixels per block along x
#define SPX   (TPX + 2*PDISP) // 24 second-pixels staged per row
#define CPAD  136             // padded channel stride in LDS (bf16): 272B, 16B aligned
#define NTHREADS 288          // 9 waves of 32: one wave per dy

__global__ __launch_bounds__(NTHREADS)
void FunctionCorrelation_17282948399394_kernel(const float* __restrict__ first,
                                               const float* __restrict__ second,
                                               float* __restrict__ out)
{
    __shared__ bf16_t sA[TPX * CPAD];          //  4,352 B
    __shared__ bf16_t sS[NDY * SPX * CPAD];    // 58,752 B

    const int xb  = blockIdx.x * TPX;
    const int y   = blockIdx.y;
    const int b   = blockIdx.z;
    const int tid = threadIdx.x;

    // ---- stage first tile: 16 px x 128 ch, fp32 -> bf16 (coalesced on px) ----
    for (int idx = tid; idx < TPX * CCH; idx += NTHREADS) {
        const int px = idx & (TPX - 1);
        const int c  = idx >> 4;
        const float v = first[(((size_t)b * CCH + c) * HH + y) * WW + (xb + px)];
        sA[px * CPAD + c] = (bf16_t)v;
    }

    // ---- stage second: 9 rows x 24 px x 128 ch, zero-padded at borders ----
    for (int idx = tid; idx < NDY * SPX * CCH; idx += NTHREADS) {
        const int px = idx % SPX;
        const int t  = idx / SPX;
        const int c  = t % CCH;
        const int r  = t / CCH;                 // dy + 4
        const int gy = y + r - PDISP;
        const int gx = xb + px - PDISP;
        float v = 0.0f;
        if ((unsigned)gy < (unsigned)HH && (unsigned)gx < (unsigned)WW)
            v = second[(((size_t)b * CCH + c) * HH + gy) * WW + gx];
        sS[(r * SPX + px) * CPAD + c] = (bf16_t)v;
    }
    __syncthreads();

    const int wave = tid >> 5;     // 0..8  == dy index r
    const int lane = tid & 31;
    const int half = lane >> 4;    // K-half select per ISA fragment layout
    const int m    = lane & 15;    // A row / B column within tile

    // A fragments for all 4 K-blocks (contiguous K order, matching B:
    // lanes<16: K = kb*32 + 0..15; lanes>=16: K = kb*32 + 16..31)
    v16bf afr[4];
#pragma unroll
    for (int kb = 0; kb < 4; ++kb) {
        const bf16_t* ap = &sA[m * CPAD + kb * 32 + half * 16];
        const v8bf lo = *(const v8bf*)ap;         // ds_load_b128
        const v8bf hi = *(const v8bf*)(ap + 8);   // ds_load_b128
#pragma unroll
        for (int i = 0; i < 8; ++i) { afr[kb][i] = lo[i]; afr[kb][8 + i] = hi[i]; }
    }

    const int r = wave;            // this wave's dy index
    v8f acc0 = {};                 // column tile 0: second px 0..15  (gx = xb-4 .. xb+11)
    v8f acc1 = {};                 // column tile 1: second px 8..23  (gx = xb+4 .. xb+19)

#pragma unroll
    for (int kb = 0; kb < 4; ++kb) {
        // B fragment (contiguous K order as A: lanes<16 hold K 0..15, lanes>=16 hold K 16..31)
        const bf16_t* bp0 = &sS[(r * SPX + 0 + m) * CPAD + kb * 32 + half * 16];
        const v8bf b0lo = *(const v8bf*)bp0;
        const v8bf b0hi = *(const v8bf*)(bp0 + 8);
        v16bf bf0;
#pragma unroll
        for (int i = 0; i < 8; ++i) { bf0[i] = b0lo[i]; bf0[8 + i] = b0hi[i]; }
        acc0 = __builtin_amdgcn_wmma_f32_16x16x32_bf16(
            false, afr[kb], false, bf0, (short)0, acc0, false, false);

        const bf16_t* bp1 = &sS[(r * SPX + 8 + m) * CPAD + kb * 32 + half * 16];
        const v8bf b1lo = *(const v8bf*)bp1;
        const v8bf b1hi = *(const v8bf*)(bp1 + 8);
        v16bf bf1;
#pragma unroll
        for (int i = 0; i < 8; ++i) { bf1[i] = b1lo[i]; bf1[8 + i] = b1hi[i]; }
        acc1 = __builtin_amdgcn_wmma_f32_16x16x32_bf16(
            false, afr[kb], false, bf1, (short)0, acc1, false, false);
    }

    // ---- band extraction + writeback ----
    // C/D layout: VGPR v holds M=v (lanes 0-15) / M=v+8 (lanes 16-31), N=lane%16.
    // tile0 serves pixels i=v (<8):   dx = lane - v - 4      -> dd = lane - v
    // tile1 serves pixels i=v+8:      dx = (lane-16) - v - 4 -> dd = lane - 16 - v
    const float scale = 1.0f / (float)CCH;
    float* obase = out + (((size_t)b * NDISP + r * NDY) * HH + y) * WW + xb;

    if (lane < 16) {
#pragma unroll
        for (int v = 0; v < 8; ++v) {
            const int dd = lane - v;
            if (dd >= 0 && dd <= 8)
                obase[(size_t)dd * HH * WW + v] = acc0[v] * scale;
        }
    } else {
#pragma unroll
        for (int v = 0; v < 8; ++v) {
            const int dd = lane - 16 - v;
            if (dd >= 0 && dd <= 8)
                obase[(size_t)dd * HH * WW + 8 + v] = acc1[v] * scale;
        }
    }
}

extern "C" void kernel_launch(void* const* d_in, const int* in_sizes, int n_in,
                              void* d_out, int out_size, void* d_ws, size_t ws_size,
                              hipStream_t stream) {
    const float* first  = (const float*)d_in[0];
    const float* second = (const float*)d_in[1];
    float* out = (float*)d_out;

    dim3 grid(WW / TPX, HH, BATCH);   // 12 x 128 x 4 = 6144 blocks
    FunctionCorrelation_17282948399394_kernel<<<grid, NTHREADS, 0, stream>>>(first, second, out);

    (void)in_sizes; (void)n_in; (void)out_size; (void)d_ws; (void)ws_size;
}
